// FeatureInspector_558345749224
// MI455X (gfx1250) — matrix operand order repaired and checked by reading, not compiled, not run on hardware
//
#include <hip/hip_runtime.h>
#include <hip/hip_bf16.h>

// ---------------------------------------------------------------------------
// FeatureInspector on MI455X (gfx1250, wave32, WMMA, async-to-LDS)
//   K1: SKA (dynamic 3x3 aggregation)            -> d_out (f32 ska)
//   K2: grouped 3x3 conv (WMMA bf16) +BN1+ReLU   -> ws t1 (bf16, [b][pix][c])
//   K3: 1x1 conv GEMM (WMMA bf16) +BN2, +=ska    -> d_out (f32)
// ---------------------------------------------------------------------------

#define Bn 8
#define Cc 256
#define Hh 128
#define Ww 128
#define Gg 8
#define GC 32          // channels per group
#define HW (Hh*Ww)     // 16384

typedef __bf16        v16bf __attribute__((ext_vector_type(16)));
typedef float         v8f   __attribute__((ext_vector_type(8)));
typedef unsigned int  v4u   __attribute__((ext_vector_type(4)));

union Frag16 { v16bf v; v4u q[2]; };

static __device__ __forceinline__ unsigned short f2bf(float f) {
    union { float f; unsigned int u; } c; c.f = f;
    unsigned int u = c.u;
    unsigned int r = (u + 0x7FFFu + ((u >> 16) & 1u)) >> 16;   // RNE
    return (unsigned short)r;
}

// Per-lane 16B async copy global -> LDS (ASYNCcnt-tracked DMA path).
static __device__ __forceinline__ void async_cp_b128(unsigned lds_addr,
                                                     unsigned long long gaddr) {
    asm volatile("global_load_async_to_lds_b128 %0, %1, off"
                 :: "v"(lds_addr), "v"(gaddr) : "memory");
}
static __device__ __forceinline__ void wait_async0() {
    asm volatile("s_wait_asynccnt 0x0" ::: "memory");
}
static __device__ __forceinline__ unsigned lds_off(const void* p) {
    // flat shared address: addr[31:0] is the LDS byte offset (ISA 10.2 aperture map)
    return (unsigned)(unsigned long long)p;
}

// ---------------------------------------------------------------------------
// Prep: fold BN scales into bf16 weights, build biases.
//   w1bf[oc][tap*32+ic]  (256 x 288),   w2bf[oc][ic] (256 x 256)
// ---------------------------------------------------------------------------
__global__ __launch_bounds__(256) void prep_kernel(
    const float* __restrict__ w1, const float* __restrict__ g1,
    const float* __restrict__ b1, const float* __restrict__ m1,
    const float* __restrict__ v1, const float* __restrict__ w2,
    const float* __restrict__ g2, const float* __restrict__ b2,
    const float* __restrict__ m2, const float* __restrict__ v2,
    unsigned short* __restrict__ w1bf, unsigned short* __restrict__ w2bf,
    float* __restrict__ bias1, float* __restrict__ bias2)
{
    int id = blockIdx.x * 256 + threadIdx.x;
    if (id < 73728) {                               // w1: 256*288
        int oc = id / 288, rem = id % 288;
        int tap = rem >> 5, ic = rem & 31;
        float s = g1[oc] * rsqrtf(v1[oc] + 1e-5f);
        // src layout (oc, ic, kh, kw) = oc*288 + ic*9 + tap
        w1bf[oc * 288 + tap * 32 + ic] = f2bf(w1[oc * 288 + ic * 9 + tap] * s);
    } else if (id < 73728 + 65536) {                // w2: 256*256
        int id2 = id - 73728;
        int oc = id2 >> 8;
        float s = g2[oc] * rsqrtf(v2[oc] + 1e-5f);
        w2bf[id2] = f2bf(w2[id2] * s);
    } else if (id < 73728 + 65536 + 256) {
        int oc = id - (73728 + 65536);
        float s = g1[oc] * rsqrtf(v1[oc] + 1e-5f);
        bias1[oc] = b1[oc] - m1[oc] * s;
    } else if (id < 73728 + 65536 + 512) {
        int oc = id - (73728 + 65536 + 256);
        float s = g2[oc] * rsqrtf(v2[oc] + 1e-5f);
        bias2[oc] = b2[oc] - m2[oc] * s;
    }
}

// ---------------------------------------------------------------------------
// K1: SKA. One block per (b, g, h); 128 threads = one pixel column each.
// ---------------------------------------------------------------------------
__global__ __launch_bounds__(128) void ska_kernel(
    const float* __restrict__ x, const float* __restrict__ dwm,
    float* __restrict__ out)
{
    int bid = blockIdx.x;
    int h = bid & 127, g = (bid >> 7) & 7, b = bid >> 10;
    int w = threadIdx.x;

    float dwgt[9];
    size_t dwbase = (((size_t)(b * Gg + g) * 9) * Hh + h) * Ww + w;
#pragma unroll
    for (int t = 0; t < 9; ++t) dwgt[t] = dwm[dwbase + (size_t)t * HW];

    for (int cl = 0; cl < GC; ++cl) {
        int c = g * GC + cl;
        const float* xp = x + (((size_t)b * Cc + c) << 14);
        __builtin_prefetch(xp + HW, 0, 1);          // global_prefetch_b8 next channel
        float acc = 0.f;
#pragma unroll
        for (int t = 0; t < 9; ++t) {
            int hh = h + t / 3 - 1;
            int ww = w + t % 3 - 1;
            float xv = 0.f;
            if ((unsigned)hh < (unsigned)Hh && (unsigned)ww < (unsigned)Ww)
                xv = xp[hh * Ww + ww];
            acc += xv * dwgt[t];
        }
        out[(((size_t)b * Cc + c) << 14) + h * Ww + w] = acc;
    }
}

// ---------------------------------------------------------------------------
// K2: grouped 3x3 conv as WMMA GEMM  (M=32 oc, K=9*32, N=128 pixels/row).
// Block = (b, vg, h), 256 threads (8 waves). 18 v_wmma per wave.
// Weights async-DMA'd to LDS; input = ska*roi -> bf16 in LDS (zero padded).
// Epilogue: +bias1, ReLU, two b128 stores/lane into t1[b][pix][c].
// ---------------------------------------------------------------------------
__global__ __launch_bounds__(256) void gconv3_kernel(
    const float* __restrict__ ska, const float* __restrict__ roi,
    const unsigned short* __restrict__ w1bf, const float* __restrict__ bias1,
    unsigned short* __restrict__ t1)
{
    __shared__ __align__(16) unsigned short inlds[3 * 130 * GC]; // [r][col][ic]
    __shared__ __align__(16) unsigned short wlds[GC * 288];      // [oc][k] 18KB

    int bid = blockIdx.x;
    int h = bid & 127, vg = (bid >> 7) & 7, b = bid >> 10;
    int tid = threadIdx.x;

    // async-DMA folded weights for this group (contiguous 18KB)
    {
        unsigned ldsw = lds_off(&wlds[0]);
        unsigned long long gw =
            (unsigned long long)(const void*)(w1bf + (size_t)vg * GC * 288);
        for (int c = tid; c < (GC * 288 * 2) / 16; c += 256)
            async_cp_b128(ldsw + c * 16, gw + (unsigned long long)c * 16);
    }
    // stage masked input rows h-1..h+1, bf16, zero padded: [r][col 0..129][ic]
    // col varies fastest across tid -> contiguous global reads of ska rows
    {
        size_t xbase = (((size_t)b * Cc + vg * GC) << 14);
        const float* roib = roi + ((size_t)b << 14);
        for (int i = tid; i < 3 * 130 * GC; i += 256) {
            int col = i % 130;
            int rest = i / 130;
            int ic = rest & 31, r = rest >> 5;
            int hh = h + r - 1, ww = col - 1;
            float v = 0.f;
            if ((unsigned)hh < (unsigned)Hh && (unsigned)ww < (unsigned)Ww)
                v = ska[xbase + ((size_t)ic << 14) + hh * Ww + ww] * roib[hh * Ww + ww];
            inlds[((r * 130 + col) << 5) + ic] = f2bf(v);
        }
    }
    wait_async0();
    __syncthreads();

    int wv = tid >> 5, lane = tid & 31;
    int w0 = wv * 16;              // N-tile base pixel
    int m = lane & 15, half = lane >> 4;

    v8f acc0 = {0.f,0.f,0.f,0.f,0.f,0.f,0.f,0.f};
    v8f acc1 = {0.f,0.f,0.f,0.f,0.f,0.f,0.f,0.f};

#pragma unroll
    for (int t = 0; t < 9; ++t) {
        const int dh = t / 3, dw = t % 3;
        Frag16 a0, a1, bf;
        const int ka = half * 8;
        {   // A fragments: rows [0..15] / [16..31] of group weights, K=t*32..
            const unsigned short* p0 = &wlds[m * 288 + t * 32];
            a0.q[0] = *(const v4u*)&p0[ka];
            a0.q[1] = *(const v4u*)&p0[16 + ka];
            const unsigned short* p1 = &wlds[(16 + m) * 288 + t * 32];
            a1.q[0] = *(const v4u*)&p1[ka];
            a1.q[1] = *(const v4u*)&p1[16 + ka];
        }
        {   // B fragment: column = pixel w0+m shifted by tap; K = 32 in-ch
            int col = w0 + m + dw;
            const unsigned short* p = &inlds[((dh * 130 + col) << 5) + half * 8];
            bf.q[0] = *(const v4u*)&p[0];
            bf.q[1] = *(const v4u*)&p[16];
        }
        acc0 = __builtin_amdgcn_wmma_f32_16x16x32_bf16(false, a0.v, false, bf.v,
                                                       (short)0, acc0, false, false);
        acc1 = __builtin_amdgcn_wmma_f32_16x16x32_bf16(false, a1.v, false, bf.v,
                                                       (short)0, acc1, false, false);
    }

    // epilogue: D layout M = r + 8*half, N = lane&15.
    // t1 layout [b][pixel][c]: 8 consecutive channels per half -> b128 stores.
    size_t pixbase = (((size_t)b << 14) + (size_t)h * Ww + (w0 + m)) * 256
                     + (size_t)vg * GC;
    union { v4u q; unsigned short u[8]; } o0, o1;
#pragma unroll
    for (int r = 0; r < 8; ++r) {
        int ocl = half * 8 + r;
        float v0 = acc0[r] + bias1[vg * GC + ocl];
        o0.u[r] = f2bf(v0 > 0.f ? v0 : 0.f);
        float v1 = acc1[r] + bias1[vg * GC + 16 + ocl];
        o1.u[r] = f2bf(v1 > 0.f ? v1 : 0.f);
    }
    *(v4u*)&t1[pixbase + half * 8] = o0.q;
    *(v4u*)&t1[pixbase + 16 + half * 8] = o1.q;
}

// ---------------------------------------------------------------------------
// K3: 1x1 conv GEMM 256x256xN (WMMA bf16). Block: 64 oc x 64 pixels.
// Both LDS tiles are contiguous 32KB async DMA copies (t1 is [b][pix][c]).
// 8 waves = 4 M-tiles x 2 N-halves; each wave: 2 accs, 16 chained v_wmma.
// Epilogue: +bias2, then d_out += y  (d_out already holds ska).
// ---------------------------------------------------------------------------
__global__ __launch_bounds__(256) void conv1x1_kernel(
    const unsigned short* __restrict__ t1, const unsigned short* __restrict__ w2bf,
    const float* __restrict__ bias2, float* __restrict__ out)
{
    __shared__ __align__(16) unsigned short alds[64 * 256];  // [ocl][k]  32KB
    __shared__ __align__(16) unsigned short blds[64 * 256];  // [pl][k]   32KB

    int ptile = blockIdx.x;           // 0..2047 (64 pixels each)
    int oc0 = blockIdx.y * 64;        // 0..3 -> oc base
    size_t p0 = (size_t)ptile * 64;
    int b = (int)(p0 >> 14);
    int pix = (int)(p0 & 16383);
    int tid = threadIdx.x;

    // async-DMA weights [oc0..oc0+63][0..255]: contiguous 32KB
    {
        unsigned la = lds_off(&alds[0]);
        unsigned long long ga =
            (unsigned long long)(const void*)(w2bf + (size_t)oc0 * 256);
        for (int c = tid; c < 2048; c += 256)
            async_cp_b128(la + c * 16, ga + (unsigned long long)c * 16);
    }
    // async-DMA activations t1[b][pix..pix+63][*]: contiguous 32KB
    {
        unsigned lb = lds_off(&blds[0]);
        unsigned long long gb =
            (unsigned long long)(const void*)(t1 + (((size_t)b << 14) + pix) * 256);
        for (int c = tid; c < 2048; c += 256)
            async_cp_b128(lb + c * 16, gb + (unsigned long long)c * 16);
    }
    wait_async0();
    __syncthreads();

    int wv = tid >> 5, lane = tid & 31;
    int mt = wv & 3, nt = wv >> 2;                 // 4 M-tiles x 2 N-halves
    int m = lane & 15, half = lane >> 4;

    v8f accA = {0.f,0.f,0.f,0.f,0.f,0.f,0.f,0.f};
    v8f accB = {0.f,0.f,0.f,0.f,0.f,0.f,0.f,0.f};
#pragma unroll
    for (int kb = 0; kb < 256; kb += 32) {
        Frag16 a, bA, bB;
        const int ka = half * 8;
        const unsigned short* pa = &alds[(mt * 16 + m) * 256 + kb];
        a.q[0] = *(const v4u*)&pa[ka];
        a.q[1] = *(const v4u*)&pa[16 + ka];
        const unsigned short* pbA = &blds[(nt * 32 + m) * 256 + kb + half * 8];
        bA.q[0] = *(const v4u*)&pbA[0];
        bA.q[1] = *(const v4u*)&pbA[16];
        const unsigned short* pbB = &blds[(nt * 32 + 16 + m) * 256 + kb + half * 8];
        bB.q[0] = *(const v4u*)&pbB[0];
        bB.q[1] = *(const v4u*)&pbB[16];
        accA = __builtin_amdgcn_wmma_f32_16x16x32_bf16(false, a.v, false, bA.v,
                                                       (short)0, accA, false, false);
        accB = __builtin_amdgcn_wmma_f32_16x16x32_bf16(false, a.v, false, bB.v,
                                                       (short)0, accB, false, false);
    }

    // epilogue: out[b][oc][pix] = ska + (acc + bias2)   (NCHW output)
#pragma unroll
    for (int r = 0; r < 8; ++r) {
        int ml = r + half * 8;
        int oc = oc0 + mt * 16 + ml;
        float bias = bias2[oc];
        size_t rowbase = (((size_t)b * Cc + oc) << 14) + pix;
        size_t iA = rowbase + nt * 32 + m;
        out[iA] += accA[r] + bias;
        size_t iB = rowbase + nt * 32 + 16 + m;
        out[iB] += accB[r] + bias;
    }
}

// ---------------------------------------------------------------------------
extern "C" void kernel_launch(void* const* d_in, const int* in_sizes, int n_in,
                              void* d_out, int out_size, void* d_ws, size_t ws_size,
                              hipStream_t stream) {
    (void)in_sizes; (void)n_in; (void)out_size; (void)ws_size;
    const float* x   = (const float*)d_in[0];
    const float* dwm = (const float*)d_in[1];
    const float* roi = (const float*)d_in[2];
    const float* w1  = (const float*)d_in[3];
    const float* g1  = (const float*)d_in[4];
    const float* b1  = (const float*)d_in[5];
    const float* m1  = (const float*)d_in[6];
    const float* v1  = (const float*)d_in[7];
    const float* w2  = (const float*)d_in[8];
    const float* g2  = (const float*)d_in[9];
    const float* b2  = (const float*)d_in[10];
    const float* m2  = (const float*)d_in[11];
    const float* v2  = (const float*)d_in[12];
    float* out = (float*)d_out;

    char* ws = (char*)d_ws;
    unsigned short* t1   = (unsigned short*)ws;                       // 64 MiB
    unsigned short* w1bf = (unsigned short*)(ws + (size_t)Bn * Cc * HW * 2);
    unsigned short* w2bf = w1bf + 73728;
    float* bias1 = (float*)(w2bf + 65536);
    float* bias2 = bias1 + 256;

    prep_kernel<<<546, 256, 0, stream>>>(w1, g1, b1, m1, v1, w2, g2, b2, m2, v2,
                                         w1bf, w2bf, bias1, bias2);
    ska_kernel<<<Bn * Gg * Hh, 128, 0, stream>>>(x, dwm, out);
    gconv3_kernel<<<Bn * Gg * Hh, 256, 0, stream>>>(out, roi, w1bf, bias1, t1);
    conv1x1_kernel<<<dim3(2048, 4), 256, 0, stream>>>(t1, w2bf, bias2, out);
}
